// InfrastructureGNN_43327630082785
// MI455X (gfx1250) — compile-verified, matrix operand order checked by reading, not run on hardware
//
#include <hip/hip_runtime.h>
#include <stdint.h>

#define HIDD 128
#define NHEADS 4
#define HCD 32

typedef __bf16 bf16_t;
typedef bf16_t v16bf __attribute__((ext_vector_type(16)));
typedef bf16_t v8bf  __attribute__((ext_vector_type(8)));
typedef float  v8f   __attribute__((ext_vector_type(8)));

// ---------- helpers ----------
__device__ __forceinline__ bf16_t f2bf(float f) {
  unsigned u = __float_as_uint(f);
  unsigned r = (u + 0x7FFFu + ((u >> 16) & 1u)) >> 16;   // round-to-nearest-even
  unsigned short s = (unsigned short)r;
  return __builtin_bit_cast(bf16_t, s);
}
// monotone float<->u32 encoding for atomicMax on float values
__device__ __forceinline__ unsigned fenc(float f) {
  unsigned u = __float_as_uint(f);
  return (u & 0x80000000u) ? ~u : (u | 0x80000000u);
}
__device__ __forceinline__ float fdec(unsigned k) {
  unsigned u = (k & 0x80000000u) ? (k ^ 0x80000000u) : ~k;
  return __uint_as_float(u);
}
__device__ __forceinline__ float lrelu(float x) { return x > 0.f ? x : 0.2f * x; }

// ---------- trivial fills ----------
__global__ void k_fill0(float* __restrict__ p, size_t n) {
  size_t i = (size_t)blockIdx.x * blockDim.x + threadIdx.x;
  size_t st = (size_t)gridDim.x * blockDim.x;
  for (; i < n; i += st) p[i] = 0.f;
}

// ---------- degree / count ----------
__global__ void k_edge_deg(const long long* __restrict__ ei, const float* __restrict__ w,
                           float* __restrict__ sumw, float* __restrict__ cnt, int E) {
  int e = blockIdx.x * blockDim.x + threadIdx.x;
  if (e >= E) return;
  int d = (int)ei[(size_t)E + e];
  atomicAdd(&sumw[d], w[e]);
  atomicAdd(&cnt[d], 1.0f);
}

__global__ void k_node_prep(const float* __restrict__ sumw, const float* __restrict__ cnt,
                            float* __restrict__ dinv, float* __restrict__ loopat, int N) {
  int n = blockIdx.x * blockDim.x + threadIdx.x;
  if (n >= N) return;
  float sw = sumw[n];
  float deg = sw + 1.0f;                       // + self-loop weight 1
  dinv[n] = (deg > 0.f) ? rsqrtf(deg) : 0.f;
  loopat[n] = sw / fmaxf(cnt[n], 1.0f);
}

// ---------- WMMA GEMM: out[N x NOUT] = A[N x KIN] * W[KIN x NOUT] (+bias) ----------
// bf16 inputs built on the fly; W staged in LDS K-major per output column so each
// lane's B fragment is two 16B ds_read_b128 loads. One wave = 16 rows x NPAD cols.
template <int KIN, int NOUT>
__global__ __launch_bounds__(256) void k_gemm(const float* __restrict__ A,
                                              const float* __restrict__ W,
                                              const float* __restrict__ bias,
                                              float* __restrict__ out, int nrows) {
  constexpr int KPAD = ((KIN + 31) / 32) * 32;
  constexpr int KSTEPS = KPAD / 32;
  constexpr int NPAD = ((NOUT + 15) / 16) * 16;
  constexpr int NT = NPAD / 16;

  __shared__ bf16_t Wl[NPAD * KPAD];             // Wl[n*KPAD + k] = W[k][n], zero-padded
  for (int i = threadIdx.x; i < NPAD * KPAD; i += 256) {
    int n = i / KPAD, k = i % KPAD;
    float v = (n < NOUT && k < KIN) ? W[(size_t)k * NOUT + n] : 0.f;
    Wl[i] = f2bf(v);
  }
  __syncthreads();

  const int wave = threadIdx.x >> 5, lane = threadIdx.x & 31;
  const int l16 = lane & 15, khalf = lane >> 4;
  const int rowA = blockIdx.x * 128 + wave * 16 + l16;   // A-fragment row for this lane
  const bool rowok = rowA < nrows;

  v8f acc[NT] = {};

#pragma unroll
  for (int ks = 0; ks < KSTEPS; ++ks) {
    const int ka = ks * 32 + khalf * 8;        // frag elems 0..7
    const int kb = ks * 32 + 16 + khalf * 8;   // frag elems 8..15
    v16bf af;
#pragma unroll
    for (int e = 0; e < 8; ++e) {
      float va = (rowok && (ka + e) < KIN) ? A[(size_t)rowA * KIN + ka + e] : 0.f;
      float vb = (rowok && (kb + e) < KIN) ? A[(size_t)rowA * KIN + kb + e] : 0.f;
      af[e] = f2bf(va);
      af[e + 8] = f2bf(vb);
    }
#pragma unroll
    for (int t = 0; t < NT; ++t) {
      const int ncol = t * 16 + l16;
      v8bf b0 = *(const v8bf*)&Wl[ncol * KPAD + ka];
      v8bf b1 = *(const v8bf*)&Wl[ncol * KPAD + kb];
      v16bf bfg;
#pragma unroll
      for (int e = 0; e < 8; ++e) { bfg[e] = b0[e]; bfg[e + 8] = b1[e]; }
      acc[t] = __builtin_amdgcn_wmma_f32_16x16x32_bf16(
          /*neg_a=*/false, af, /*neg_b=*/false, bfg,
          /*c_mod=*/(short)0, acc[t], /*reuse_a=*/false, /*reuse_b=*/false);
    }
  }

  // C/D layout: VGPR i -> M = i + 8*khalf, N = lane&15
  const int rbase = blockIdx.x * 128 + wave * 16 + khalf * 8;
#pragma unroll
  for (int t = 0; t < NT; ++t) {
    const int col = t * 16 + l16;
    if (col < NOUT) {
      const float bv = bias ? bias[col] : 0.f;
#pragma unroll
      for (int i = 0; i < 8; ++i) {
        const int r = rbase + i;
        if (r < nrows) out[(size_t)r * NOUT + col] = acc[t][i] + bv;
      }
    }
  }
}

// ---------- GCN edge aggregation: acc[dst] += dinv[src]*w*dinv[dst] * h[src] ----------
__global__ void k_gcn_agg(const float* __restrict__ h, const long long* __restrict__ ei,
                          const float* __restrict__ w, const float* __restrict__ dinv,
                          float* __restrict__ acc, int E) {
  const int wave = threadIdx.x >> 5, lane = threadIdx.x & 31;
  const long e = (long)blockIdx.x * 8 + wave;
  if (e >= E) return;
  const int s = (int)ei[e], d = (int)ei[(long)E + e];
  const float nrm = dinv[s] * w[e] * dinv[d];
  float4 v = *((const float4*)(h + (size_t)s * HIDD) + lane);
  float* op = acc + (size_t)d * HIDD + lane * 4;
  atomicAdd(op + 0, nrm * v.x);
  atomicAdd(op + 1, nrm * v.y);
  atomicAdd(op + 2, nrm * v.z);
  atomicAdd(op + 3, nrm * v.w);
}

// ---------- GCN finalize: acc = relu(acc + dinv^2*h + b (+resid)) ----------
__global__ void k_gcn_fin(float* __restrict__ acc, const float* __restrict__ h,
                          const float* __restrict__ dinv, const float* __restrict__ b,
                          const float* __restrict__ resid, size_t total) {
  size_t i = (size_t)blockIdx.x * blockDim.x + threadIdx.x;
  if (i >= total) return;
  int n = (int)(i >> 7), f = (int)(i & 127);
  float di = dinv[n];
  float v = acc[i] + di * di * h[i] + b[f];
  if (resid) v += resid[i];
  acc[i] = fmaxf(v, 0.f);
}

// ---------- GAT: ce[h] = sum_c We[h*32+c]*att_e[h*32+c] ----------
__global__ void k_ce(const float* __restrict__ We, const float* __restrict__ att_e,
                     float* __restrict__ ce) {
  int h = threadIdx.x;
  if (h >= NHEADS) return;
  float s = 0.f;
  for (int c = 0; c < HCD; ++c) s += We[h * HCD + c] * att_e[h * HCD + c];
  ce[h] = s;
}

// ---------- GAT: per-node head dot products (wave per node) ----------
__global__ void k_att_nodes(const float* __restrict__ hg, const float* __restrict__ asrc,
                            const float* __restrict__ adst, float* __restrict__ a_s,
                            float* __restrict__ a_d, int N) {
  const int wave = threadIdx.x >> 5, lane = threadIdx.x & 31;
  const int n = blockIdx.x * 8 + wave;
  if (n >= N) return;
  const float* hp = hg + (size_t)n * HIDD;
#pragma unroll
  for (int h = 0; h < NHEADS; ++h) {
    float x = hp[h * HCD + lane];
    float vs = x * asrc[h * HCD + lane];
    float vd = x * adst[h * HCD + lane];
#pragma unroll
    for (int off = 16; off > 0; off >>= 1) {
      vs += __shfl_xor(vs, off, 32);
      vd += __shfl_xor(vd, off, 32);
    }
    if (lane == 0) { a_s[n * NHEADS + h] = vs; a_d[n * NHEADS + h] = vd; }
  }
}

// ---------- GAT: self-loop logits seed the segment max ----------
__global__ void k_alpha_self(const float* __restrict__ a_s, const float* __restrict__ a_d,
                             const float* __restrict__ loopat, const float* __restrict__ ce,
                             float* __restrict__ aself, unsigned* __restrict__ amax, int N) {
  int i = blockIdx.x * blockDim.x + threadIdx.x;
  if (i >= N * NHEADS) return;
  int n = i / NHEADS, h = i % NHEADS;
  float a = lrelu(a_s[i] + a_d[i] + loopat[n] * ce[h]);
  aself[i] = a;
  amax[i] = fenc(a);   // every node has a self loop -> valid init for segment max
}

// ---------- GAT: edge logits + atomic segment max ----------
__global__ void k_alpha_edges(const long long* __restrict__ ei, const float* __restrict__ w,
                              const float* __restrict__ a_s, const float* __restrict__ a_d,
                              const float* __restrict__ ce, float* __restrict__ alpha,
                              unsigned* __restrict__ amax, int E) {
  int e = blockIdx.x * blockDim.x + threadIdx.x;
  if (e >= E) return;
  int s = (int)ei[e], d = (int)ei[(long)E + e];
  float we = w[e];
#pragma unroll
  for (int h = 0; h < NHEADS; ++h) {
    float a = lrelu(a_s[s * NHEADS + h] + a_d[d * NHEADS + h] + we * ce[h]);
    alpha[(size_t)e * NHEADS + h] = a;
    atomicMax(&amax[d * NHEADS + h], fenc(a));
  }
}

// ---------- GAT: decode max, self-loop exp seeds denom ----------
__global__ void k_softmax_nodes(unsigned* __restrict__ amax_enc, float* __restrict__ aself,
                                float* __restrict__ denom, int N) {
  int i = blockIdx.x * blockDim.x + threadIdx.x;
  if (i >= N * NHEADS) return;
  float m = fdec(amax_enc[i]);
  ((float*)amax_enc)[i] = m;           // buffer becomes plain float amax
  float ex = __expf(aself[i] - m);
  aself[i] = ex;                        // aself now holds ex_self
  denom[i] = ex;
}

// ---------- GAT: edge exp + atomic denom ----------
__global__ void k_softmax_edges(float* __restrict__ alpha, const float* __restrict__ amax,
                                float* __restrict__ denom, const long long* __restrict__ ei,
                                int E) {
  int e = blockIdx.x * blockDim.x + threadIdx.x;
  if (e >= E) return;
  int d = (int)ei[(long)E + e];
#pragma unroll
  for (int h = 0; h < NHEADS; ++h) {
    float ex = __expf(alpha[(size_t)e * NHEADS + h] - amax[d * NHEADS + h]);
    alpha[(size_t)e * NHEADS + h] = ex;   // alpha now holds ex
    atomicAdd(&denom[d * NHEADS + h], ex);
  }
}

// ---------- GAT: weighted aggregation (wave per edge, lane = channel) ----------
__global__ void k_gat_agg(const float* __restrict__ hg, const long long* __restrict__ ei,
                          const float* __restrict__ ex, const float* __restrict__ denom,
                          float* __restrict__ acc, int E) {
  const int wave = threadIdx.x >> 5, lane = threadIdx.x & 31;
  const long e = (long)blockIdx.x * 8 + wave;
  if (e >= E) return;
  const int s = (int)ei[e], d = (int)ei[(long)E + e];
#pragma unroll
  for (int h = 0; h < NHEADS; ++h) {
    float a = ex[(size_t)e * NHEADS + h] / denom[d * NHEADS + h];
    atomicAdd(&acc[(size_t)d * HIDD + h * HCD + lane],
              a * hg[(size_t)s * HIDD + h * HCD + lane]);
  }
}

// ---------- GAT finalize: x2 = relu(acc + a_self*hg + bg + x1) ----------
__global__ void k_gat_fin(float* __restrict__ acc, const float* __restrict__ hg,
                          const float* __restrict__ exself, const float* __restrict__ denom,
                          const float* __restrict__ bg, const float* __restrict__ x1,
                          size_t total) {
  size_t i = (size_t)blockIdx.x * blockDim.x + threadIdx.x;
  if (i >= total) return;
  int n = (int)(i >> 7), f = (int)(i & 127), h = f >> 5;
  float a = exself[n * NHEADS + h] / denom[n * NHEADS + h];
  float v = acc[i] + a * hg[i] + bg[f] + x1[i];
  acc[i] = fmaxf(v, 0.f);
}

// ---------- host ----------
extern "C" void kernel_launch(void* const* d_in, const int* in_sizes, int n_in,
                              void* d_out, int out_size, void* d_ws, size_t ws_size,
                              hipStream_t stream) {
  (void)n_in; (void)out_size; (void)ws_size;
  const float*     x       = (const float*)d_in[0];
  const long long* ei      = (const long long*)d_in[1];
  const float*     ew      = (const float*)d_in[2];
  const float*     W1      = (const float*)d_in[3];
  const float*     b1      = (const float*)d_in[4];
  const float*     Wg      = (const float*)d_in[5];
  const float*     att_src = (const float*)d_in[6];
  const float*     att_dst = (const float*)d_in[7];
  const float*     We      = (const float*)d_in[8];
  const float*     att_e   = (const float*)d_in[9];
  const float*     bg      = (const float*)d_in[10];
  const float*     W3      = (const float*)d_in[11];
  const float*     b3      = (const float*)d_in[12];
  const float*     Wo      = (const float*)d_in[13];
  const float*     bo      = (const float*)d_in[14];
  float* out = (float*)d_out;

  const int N = in_sizes[0] / 24;   // IN_DIM = 24
  const int E = in_sizes[2];

  char* ws = (char*)d_ws;
  size_t off = 0;
  auto alloc = [&](size_t bytes) -> char* {
    char* p = ws + off;
    off += (bytes + 255) & ~(size_t)255;
    return p;
  };
  const size_t NB = (size_t)N * HIDD * sizeof(float);
  float* x1     = (float*)alloc(NB);                         // x1 (residual, live to end)
  float* hbuf   = (float*)alloc(NB);                         // h1 / hg / h3
  float* x2     = (float*)alloc(NB);                         // GAT acc -> x2 -> GCN2 acc -> x3
  float* sumw   = (float*)alloc((size_t)N * 4);
  float* cnt    = (float*)alloc((size_t)N * 4);
  float* dinv   = (float*)alloc((size_t)N * 4);
  float* loopat = (float*)alloc((size_t)N * 4);
  float* a_s    = (float*)alloc((size_t)N * NHEADS * 4);
  float* a_d    = (float*)alloc((size_t)N * NHEADS * 4);
  float* aself  = (float*)alloc((size_t)N * NHEADS * 4);     // alpha_self -> ex_self
  float* amax   = (float*)alloc((size_t)N * NHEADS * 4);     // encoded u32 -> float
  float* denom  = (float*)alloc((size_t)N * NHEADS * 4);
  float* ce     = (float*)alloc(256);
  float* alpha  = (float*)alloc((size_t)E * NHEADS * 4);     // alpha -> ex

  const int T = 256;
  const int gN   = (N + T - 1) / T;
  const int gE   = (E + T - 1) / T;
  const int gEw  = (E + 7) / 8;               // wave per edge
  const int gNw  = (N + 7) / 8;               // wave per node
  const int gRow = (N + 127) / 128;           // GEMM: 128 rows/block
  const size_t totNF = (size_t)N * HIDD;
  const int gNF  = (int)((totNF + T - 1) / T);
  const int gNH  = (N * NHEADS + T - 1) / T;

  // degrees / counts
  k_fill0<<<1024, T, 0, stream>>>(sumw, (size_t)N);
  k_fill0<<<1024, T, 0, stream>>>(cnt, (size_t)N);
  k_edge_deg<<<gE, T, 0, stream>>>(ei, ew, sumw, cnt, E);
  k_node_prep<<<gN, T, 0, stream>>>(sumw, cnt, dinv, loopat, N);

  // ---- GCN layer 1 ----
  k_gemm<24, 128><<<gRow, 256, 0, stream>>>(x, W1, nullptr, hbuf, N);
  k_fill0<<<2048, T, 0, stream>>>(x1, totNF);
  k_gcn_agg<<<gEw, 256, 0, stream>>>(hbuf, ei, ew, dinv, x1, E);
  k_gcn_fin<<<gNF, T, 0, stream>>>(x1, hbuf, dinv, b1, nullptr, totNF);

  // ---- GAT layer ----
  k_gemm<128, 128><<<gRow, 256, 0, stream>>>(x1, Wg, nullptr, hbuf, N);
  k_ce<<<1, 32, 0, stream>>>(We, att_e, ce);
  k_att_nodes<<<gNw, 256, 0, stream>>>(hbuf, att_src, att_dst, a_s, a_d, N);
  k_alpha_self<<<gNH, T, 0, stream>>>(a_s, a_d, loopat, ce, aself, (unsigned*)amax, N);
  k_alpha_edges<<<gE, T, 0, stream>>>(ei, ew, a_s, a_d, ce, alpha, (unsigned*)amax, E);
  k_softmax_nodes<<<gNH, T, 0, stream>>>((unsigned*)amax, aself, denom, N);
  k_softmax_edges<<<gE, T, 0, stream>>>(alpha, amax, denom, ei, E);
  k_fill0<<<2048, T, 0, stream>>>(x2, totNF);
  k_gat_agg<<<gEw, 256, 0, stream>>>(hbuf, ei, alpha, denom, x2, E);
  k_gat_fin<<<gNF, T, 0, stream>>>(x2, hbuf, aself, denom, bg, x1, totNF);

  // ---- GCN layer 2 (x2 consumed by GEMM first, then reused as accumulator) ----
  k_gemm<128, 128><<<gRow, 256, 0, stream>>>(x2, W3, nullptr, hbuf, N);
  k_fill0<<<2048, T, 0, stream>>>(x2, totNF);
  k_gcn_agg<<<gEw, 256, 0, stream>>>(hbuf, ei, ew, dinv, x2, E);
  k_gcn_fin<<<gNF, T, 0, stream>>>(x2, hbuf, dinv, b3, x1, totNF);

  // ---- output projection ----
  k_gemm<128, 12><<<gRow, 256, 0, stream>>>(x2, Wo, bo, out, N);
}
